// Attention_57526791963151
// MI455X (gfx1250) — matrix operand order repaired and checked by reading, not compiled, not run on hardware
//
#include <hip/hip_runtime.h>
#include <hip/hip_bf16.h>
#include <math.h>
#include <stdint.h>

// ---------------------------------------------------------------------------
// MI455X (gfx1250) causal attention:
//   k0: fp32 -> bf16 conversion pre-pass for x and the three weight matrices
//   k1: QKV projection (bf16 WMMA, fp32 acc), 16x128 strip per wave,
//       x fragments register-resident; V stored transposed.
//   k2: flash attention, 4 waves/WG sharing double-buffered K/V blocks staged
//       via GLOBAL_LOAD_ASYNC_TO_LDS_B128; O accumulators in registers;
//       full-wave online softmax with cross-half shuffle reduction.
// ---------------------------------------------------------------------------

typedef __bf16 bf16_t;
typedef bf16_t v16bf __attribute__((ext_vector_type(16)));
typedef bf16_t v8bf  __attribute__((ext_vector_type(8)));
typedef float  v8f   __attribute__((ext_vector_type(8)));

#define N_TOK  8192
#define DIM    512
#define WAVES  4           // waves per flash-attn workgroup (64 query rows)
#define OTILES 8           // 16x16 output tiles per projection wave (128 cols)

// ---- WMMA fragment helpers -------------------------------------------------
// 16-bit A-matrix 16x32 layout (m = lane&15, hi = lane>>4):
//   elements 0..7 -> K = hi*16 + i ; elements 8..15 -> K = hi*16 + 8 + i (as loaded here)
__device__ __forceinline__ v16bf load_a_frag(const bf16_t* p0, const bf16_t* p1) {
  v8bf lo = *(const v8bf*)p0;
  v8bf hi = *(const v8bf*)p1;
  v16bf r;
#pragma unroll
  for (int i = 0; i < 8; ++i) { r[i] = lo[i]; r[i + 8] = hi[i]; }
  return r;
}
// 16-bit B-matrix 32x16 layout (n = lane&15, hi = lane>>4): elem i -> K = hi*16+i
__device__ __forceinline__ v16bf load_b_frag(const bf16_t* p) {
  return *(const v16bf*)p;
}
__device__ __forceinline__ v8f wmma_bf16(v16bf a, v16bf b, v8f c) {
  return __builtin_amdgcn_wmma_f32_16x16x32_bf16(false, a, false, b,
                                                 (short)0, c, false, false);
}

// ---- CDNA5 async global->LDS copy (ASYNCcnt path) --------------------------
__device__ __forceinline__ void async_b128_to_lds(void* lds_ptr, unsigned gl_off,
                                                  uint64_t gl_base) {
  // Flat LDS pointers carry the LDS byte offset in addr[31:0].
  unsigned lds_off = (unsigned)(uintptr_t)lds_ptr;
  asm volatile("global_load_async_to_lds_b128 %0, %1, %2 offset:0"
               :: "v"(lds_off), "v"(gl_off), "s"(gl_base) : "memory");
}
__device__ __forceinline__ void wait_async0() {
  asm volatile("s_wait_asynccnt 0x0" ::: "memory");
}

// ---------------------------------------------------------------------------
// Kernel 0: elementwise fp32 -> bf16 (8 elements per thread, vector ld/st)
// ---------------------------------------------------------------------------
__global__ void cvt_f32_bf16(const float* __restrict__ in,
                             bf16_t* __restrict__ out, int nelem) {
  const int i0 = (blockIdx.x * blockDim.x + threadIdx.x) * 8;
  if (i0 >= nelem) return;
  const float* p = in + i0;
  v8bf v;
#pragma unroll
  for (int k = 0; k < 8; ++k) v[k] = (bf16_t)p[k];
  *(v8bf*)(out + i0) = v;
}

// ---------------------------------------------------------------------------
// Kernel 1: y = x @ W^T + b  (bf16 in -> bf16 out, fp32 accumulate).
// One wave computes a 16-token x 128-feature strip; x A-fragments loaded once.
// transposed==0: out[token][o] row-major [N,DIM]  (Q, K)
// transposed==1: out[o][token]          [DIM,N]   (V^T)
// ---------------------------------------------------------------------------
__global__ void __launch_bounds__(32, 1)
qkv_proj(const bf16_t* __restrict__ x, const bf16_t* __restrict__ W,
         const float* __restrict__ b, bf16_t* __restrict__ out, int transposed) {
  const int lane = threadIdx.x;
  const int m  = lane & 15;
  const int hi = lane >> 4;
  const int n  = m;
  const int o0base = blockIdx.x * (16 * OTILES);
  const int n0     = blockIdx.y * 16;

  v16bf xa[DIM / 32];                       // 16x512 token tile, registers
#pragma unroll
  for (int t = 0; t < DIM / 32; ++t) {
    const bf16_t* p0 = x + (size_t)(n0 + m) * DIM + t * 32 + hi * 16;
    xa[t] = load_a_frag(p0, p0 + 8);
  }

#pragma unroll
  for (int ot = 0; ot < OTILES; ++ot) {
    const int o0 = o0base + ot * 16;
    v8f acc;
    const float bias = b[o0 + n];
#pragma unroll
    for (int j = 0; j < 8; ++j) acc[j] = bias;

#pragma unroll
    for (int t = 0; t < DIM / 32; ++t) {
      // B[kk][o] = W[o0+n][t*32+kk] (contiguous along d)
      acc = wmma_bf16(xa[t], load_b_frag(W + (size_t)(o0 + n) * DIM + t * 32 + hi * 16), acc);
    }

    if (!transposed) {
#pragma unroll
      for (int j = 0; j < 8; ++j)
        out[(size_t)(n0 + hi * 8 + j) * DIM + o0 + n] = (bf16_t)acc[j];
    } else {
      v8bf v;
#pragma unroll
      for (int j = 0; j < 8; ++j) v[j] = (bf16_t)acc[j];
      *(v8bf*)(out + (size_t)(o0 + n) * N_TOK + n0 + hi * 8) = v;
    }
  }
}

// ---------------------------------------------------------------------------
// Kernel 2: flash attention. 4 waves/WG; wave w owns query rows
// [blockIdx.x*64 + w*16, +16). K/V 32-key blocks double-buffered in LDS via
// async copies; all waves cooperate on staging and share the data.
// ---------------------------------------------------------------------------
__global__ void __launch_bounds__(WAVES * 32, 1)
flash_attn(const bf16_t* __restrict__ Q, const bf16_t* __restrict__ K,
           const bf16_t* __restrict__ Vt, float* __restrict__ out) {
  __shared__ bf16_t sK[2][32 * DIM];     // 2 x 32KB: 32 keys x 512 dims
  __shared__ bf16_t sV[2][DIM * 32];     // 2 x 32KB: 512 dims x 32 keys (V^T)
  __shared__ float  sS[WAVES][16 * 32];
  __shared__ bf16_t sP[WAVES][16 * 32];
  __shared__ float  sAlpha[WAVES][16];
  __shared__ float  sLinv[WAVES][16];

  const int tid  = threadIdx.x;
  const int wave = tid >> 5;
  const int lane = tid & 31;
  const int m  = lane & 15;
  const int hi = lane >> 4;
  const int n  = m;
  const int q0 = blockIdx.x * (WAVES * 16) + wave * 16;
  const int kmax = blockIdx.x * (WAVES * 16) + WAVES * 16;  // WG causal bound
  const float scale = 0.044194173824159216f;                 // 1/sqrt(512)

  // ---- cooperative async staging of one 32-key K + V^T block -------------
  auto stage = [&](int buf, int key0) {
    const uint64_t kbase = (uint64_t)(uintptr_t)K + (uint64_t)key0 * DIM * 2;
    const uint64_t vbase = (uint64_t)(uintptr_t)Vt + (uint64_t)key0 * 2;
#pragma unroll
    for (int r = 0; r < 16; ++r) {               // 2048 16B chunks / 128 thr
      const int c   = tid + r * 128;
      const int row = c >> 6;                    // key row 0..31
      const int col = c & 63;                    // 16B chunk within 1KB row
      async_b128_to_lds(&sK[buf][c * 8], (unsigned)(row * DIM * 2 + col * 16),
                        kbase);
    }
#pragma unroll
    for (int r = 0; r < 16; ++r) {
      const int c = tid + r * 128;
      const int d    = c >> 2;                   // dim row 0..511
      const int part = c & 3;                    // 16B chunk within 64B row
      async_b128_to_lds(&sV[buf][c * 8],
                        (unsigned)(d * N_TOK * 2 + part * 16), vbase);
    }
  };

  // ---- per-wave state ------------------------------------------------------
  v16bf qa[DIM / 32];                            // whole 16x512 Q tile, regs
#pragma unroll
  for (int t = 0; t < DIM / 32; ++t) {
    const bf16_t* p0 = Q + (size_t)(q0 + m) * DIM + t * 32 + hi * 16;
    qa[t] = load_a_frag(p0, p0 + 8);
  }
  v8f o[DIM / 16];                               // 16x512 fp32 O tile, regs
#pragma unroll
  for (int t = 0; t < DIM / 16; ++t)
#pragma unroll
    for (int j = 0; j < 8; ++j) o[t][j] = 0.0f;

  // Running stats per query row, replicated in both half-waves:
  // lane (m, hi) tracks row m (lanes m and m+16 hold identical values).
  float m_i = -3.0e38f;
  float l_i = 0.0f;

  stage(0, 0);                                   // prologue: block 0
  int cur = 0;
  for (int key0 = 0; key0 < kmax; key0 += 32, cur ^= 1) {
    wait_async0();                               // our staging copies landed
    __syncthreads();                             // everyone's copies visible
    if (key0 + 32 < kmax) stage(cur ^ 1, key0 + 32);  // overlap next block

    if (key0 <= q0 + 15) {                       // causal: block relevant?
      // ---- S = Q K^T (two 16x16 tiles over 32 keys) ----
      v8f s0, s1;
#pragma unroll
      for (int j = 0; j < 8; ++j) { s0[j] = 0.0f; s1[j] = 0.0f; }
#pragma unroll
      for (int t = 0; t < DIM / 32; ++t) {
        s0 = wmma_bf16(qa[t], load_b_frag(&sK[cur][n * DIM + t * 32 + hi * 16]), s0);
        s1 = wmma_bf16(qa[t], load_b_frag(&sK[cur][(16 + n) * DIM + t * 32 + hi * 16]), s1);
      }
#pragma unroll
      for (int j = 0; j < 8; ++j) {
        const int row = hi * 8 + j;
        sS[wave][row * 32 + n]      = s0[j];
        sS[wave][row * 32 + 16 + n] = s1[j];
      }
      __builtin_amdgcn_wave_barrier();           // same-wave DS ordering

      // ---- online softmax, all 32 lanes: row m, key half hi ----
      {
        const int q = q0 + m;
        float sv[16];
        float mb = -3.0e38f;
#pragma unroll
        for (int i = 0; i < 16; ++i) {
          const int kk = hi * 16 + i;
          float s = sS[wave][m * 32 + kk] * scale;
          if (key0 + kk > q) s = -3.0e38f;       // causal mask
          sv[i] = s;
          mb = fmaxf(mb, s);
        }
        mb = fmaxf(mb, __shfl_xor(mb, 16));      // combine the two key halves
        const float mnew  = fmaxf(m_i, mb);
        const float alpha = __expf(m_i - mnew);
        float lsum = 0.0f;
#pragma unroll
        for (int i = 0; i < 16; ++i) {
          const float p = __expf(sv[i] - mnew);
          lsum += p;
          sP[wave][m * 32 + hi * 16 + i] = (bf16_t)p;
        }
        lsum += __shfl_xor(lsum, 16);
        m_i = mnew;
        l_i = l_i * alpha + lsum;
        if (lane < 16) sAlpha[wave][lane] = alpha;
      }
      __builtin_amdgcn_wave_barrier();

      // ---- O = alpha*O + P @ V  (register accumulators) ----
      const bf16_t* pp = &sP[wave][m * 32 + hi * 16];
      const v16bf pa = load_a_frag(pp, pp + 8);
      float av[8];
      bool need_rescale = false;
#pragma unroll
      for (int j = 0; j < 8; ++j) {
        av[j] = sAlpha[wave][hi * 8 + j];
        need_rescale |= (av[j] != 1.0f);
      }
      // wave-uniform skip: alpha==1 for every row when no running max moved
      if (__any(need_rescale)) {
#pragma unroll
        for (int t = 0; t < DIM / 16; ++t)
#pragma unroll
          for (int j = 0; j < 8; ++j) o[t][j] *= av[j];
      }
#pragma unroll
      for (int t = 0; t < DIM / 16; ++t) {
        const v16bf vb = load_b_frag(&sV[cur][(t * 16 + n) * 32 + hi * 16]);
        o[t] = wmma_bf16(pa, vb, o[t]);
      }
    }
    __syncthreads();                             // done reading buf `cur`
  }

  // ---- finalize: divide by softmax denominator, write fp32 output ---------
  if (lane < 16) sLinv[wave][lane] = 1.0f / l_i;
  __builtin_amdgcn_wave_barrier();
  float lv[8];
#pragma unroll
  for (int j = 0; j < 8; ++j) lv[j] = sLinv[wave][hi * 8 + j];
#pragma unroll
  for (int t = 0; t < DIM / 16; ++t)
#pragma unroll
    for (int j = 0; j < 8; ++j)
      out[(size_t)(q0 + hi * 8 + j) * DIM + t * 16 + n] = o[t][j] * lv[j];
}

// ---------------------------------------------------------------------------
extern "C" void kernel_launch(void* const* d_in, const int* in_sizes, int n_in,
                              void* d_out, int out_size, void* d_ws, size_t ws_size,
                              hipStream_t stream) {
  (void)in_sizes; (void)n_in; (void)out_size; (void)ws_size;
  const float* x  = (const float*)d_in[0];
  // d_in[1] = additive causal mask: applied analytically, not read
  const float* Wq = (const float*)d_in[2];
  const float* bq = (const float*)d_in[3];
  const float* Wk = (const float*)d_in[4];
  const float* bk = (const float*)d_in[5];
  const float* Wv = (const float*)d_in[6];
  const float* bv = (const float*)d_in[7];
  float* out = (float*)d_out;

  // Workspace: Q, K [N,D]; V^T [D,N]; xb [N,D]; Wqb/Wkb/Wvb [D,D] (all bf16)
  bf16_t* Qb  = (bf16_t*)d_ws;
  bf16_t* Kb  = Qb  + (size_t)N_TOK * DIM;
  bf16_t* Vt  = Kb  + (size_t)N_TOK * DIM;
  bf16_t* xb  = Vt  + (size_t)N_TOK * DIM;
  bf16_t* Wqb = xb  + (size_t)N_TOK * DIM;
  bf16_t* Wkb = Wqb + (size_t)DIM * DIM;
  bf16_t* Wvb = Wkb + (size_t)DIM * DIM;

  const int nx = N_TOK * DIM, nw = DIM * DIM;
  cvt_f32_bf16<<<nx / (8 * 256), 256, 0, stream>>>(x,  xb,  nx);
  cvt_f32_bf16<<<nw / (8 * 256), 256, 0, stream>>>(Wq, Wqb, nw);
  cvt_f32_bf16<<<nw / (8 * 256), 256, 0, stream>>>(Wk, Wkb, nw);
  cvt_f32_bf16<<<nw / (8 * 256), 256, 0, stream>>>(Wv, Wvb, nw);

  dim3 pgrid(DIM / (16 * OTILES), N_TOK / 16);
  qkv_proj<<<pgrid, 32, 0, stream>>>(xb, Wqb, bq, Qb, 0);
  qkv_proj<<<pgrid, 32, 0, stream>>>(xb, Wkb, bk, Kb, 0);
  qkv_proj<<<pgrid, 32, 0, stream>>>(xb, Wvb, bv, Vt, 1);

  flash_attn<<<N_TOK / (WAVES * 16), WAVES * 32, 0, stream>>>(Qb, Kb, Vt, out);
}
